// ToyModel_21715354648702
// MI455X (gfx1250) — compile-verified, matrix operand order checked by reading, not run on hardware
//
#include <hip/hip_runtime.h>
#include <hip/hip_bf16.h>

// ---------------------------------------------------------------------------
// MoE FFN for MI455X (gfx1250, wave32, WMMA), v6.
// Prep:   convert x -> bf16; convert+transpose W_e, W_out -> bf16 [n][k].
// Route:  histogram / 128-padded scan / scatter (tokens grouped by expert).
// Phase1: gathered GEMM x@W_e + b_e, ReLU -> h (bf16, gathered order).
// Phase2: GEMM h@W_out + b_out, scattered back to token order (f32 out).
// GEMM: 128x128 WG tile, 256 threads (8 waves), 64x32 wave tile,
//       double-buffered LDS pipeline, compact (non-unrolled) K loop.
//       ~150 VGPRs/wave -> no scratch spills (v5 spilled with 64x64 tiles).
// ---------------------------------------------------------------------------

#define NB     8
#define NS     2048
#define ND     1024                 // D (model dim)
#define NH     4096                 // H (hidden)
#define NE     8
#define NTOK   (NB*NS)              // 16384 tokens
#define MTILES 136                  // 128-row tiles covering 16384 + 8*127
#define MAXROWS (MTILES*128)        // 17408
#define SENTINEL 0xFFFFFFFFu

#define TILE_M 128
#define TILE_N 128
#define TILE_K 32
#define APITCH 80                   // bytes per 32-bf16 row (banking pad)
#define BPITCH 80

// workspace layout (bytes)
#define IDX_OFF    128u                            // u32 idx [MAXROWS]
#define ZROW_OFF   69760ull                        // 2KB zero bf16 row
#define XB_OFF     131072ull                       // bf16 x   [NTOK][ND]    32MB
#define WEB_OFF    (XB_OFF + 33554432ull)          // bf16 WeT [E][NH][ND]   64MB
#define WOUTB_OFF  (WEB_OFF + 67108864ull)         // bf16 WoT [ND][NH]       8MB
#define H_OFF      (WOUTB_OFF + 8388608ull)        // bf16 h [MAXROWS][NH] 142.6MB

typedef __attribute__((ext_vector_type(16))) __bf16 v16bf;
typedef __attribute__((ext_vector_type(8)))  float  v8f;

__device__ inline unsigned short f2bf(float f) {           // RNE f32->bf16
    unsigned u = __float_as_uint(f);
    u += 0x7FFFu + ((u >> 16) & 1u);
    return (unsigned short)(u >> 16);
}

__device__ inline v16bf ldfrag(const unsigned char* p0, const unsigned char* p1) {
    union { uint4 q[2]; v16bf v; } u;
    u.q[0] = *(const uint4*)p0;                            // ds_load_b128
    u.q[1] = *(const uint4*)p1;                            // ds_load_b128
    return u.v;
}

// Double-buffered GEMM mainloop. ga/gb: this thread's half-row (32B) staging
// source in A/B (bf16, k-contiguous rows). arow/brow: its LDS staging slots.
// acc = 64x32 wave tile (4 M-subtiles x 2 N-subtiles).
__device__ inline void gemm_mainloop(
    const unsigned short* __restrict__ ga, const unsigned short* __restrict__ gb,
    unsigned K, unsigned char* smA, unsigned char* smB,
    unsigned char* arow, unsigned char* brow,
    unsigned m16, unsigned half, unsigned mrow, unsigned ncol,
    v8f acc[4][2])
{
    // Hide K from the optimizer (stays in an SGPR) so the K-loop is NOT fully
    // unrolled -- keeps the hot loop small enough for the instruction cache.
    asm volatile("" : "+s"(K));
    const unsigned abuf = TILE_M * APITCH;                 // second-buffer offset

    uint4 ra0 = *(const uint4*)(ga);
    uint4 ra1 = *(const uint4*)(ga + 8u);
    uint4 rb0 = *(const uint4*)(gb);
    uint4 rb1 = *(const uint4*)(gb + 8u);
    *(uint4*)(arow)       = ra0;
    *(uint4*)(arow + 16u) = ra1;
    *(uint4*)(brow)       = rb0;
    *(uint4*)(brow + 16u) = rb1;

    const unsigned nsteps = K / TILE_K;
    #pragma unroll 2
    for (unsigned kt = 0; kt < nsteps; ++kt) {
        __syncthreads();
        const unsigned cur = (kt & 1u) * abuf;
        const unsigned nxt = abuf - cur;
        const bool more = (kt + 1u < nsteps);
        const unsigned kn = (kt + 1u) * TILE_K;
        if (more) {                                        // prefetch next K-tile
            ra0 = *(const uint4*)(ga + kn);
            ra1 = *(const uint4*)(ga + kn + 8u);
            rb0 = *(const uint4*)(gb + kn);
            rb1 = *(const uint4*)(gb + kn + 8u);
        }
        v16bf af[4];
        #pragma unroll
        for (int mi = 0; mi < 4; ++mi) {
            const unsigned char* pa =
                smA + cur + (mrow + (unsigned)mi * 16u + m16) * APITCH + half * 16u;
            af[mi] = ldfrag(pa, pa + 32);
        }
        #pragma unroll
        for (int j = 0; j < 2; ++j) {
            const unsigned char* pb =
                smB + cur + (ncol + (unsigned)j * 16u + m16) * BPITCH + half * 16u;
            v16bf bf = ldfrag(pb, pb + 32);
            #pragma unroll
            for (int mi = 0; mi < 4; ++mi)
                acc[mi][j] = __builtin_amdgcn_wmma_f32_16x16x32_bf16(
                    false, af[mi], false, bf, (short)0, acc[mi][j], false, false);
        }
        if (more) {                                        // stage into other buffer
            *(uint4*)(arow + nxt)       = ra0;
            *(uint4*)(arow + nxt + 16u) = ra1;
            *(uint4*)(brow + nxt)       = rb0;
            *(uint4*)(brow + nxt + 16u) = rb1;
        }
    }
}

// ------------------------------ routing ------------------------------------
__global__ void k_init(unsigned* counts, unsigned* cursors, unsigned* idx,
                       unsigned* zrow) {
    unsigned i = blockIdx.x * blockDim.x + threadIdx.x;
    if (i < MAXROWS) idx[i] = SENTINEL;
    if (i < NE) { counts[i] = 0u; cursors[i] = 0u; }
    if (i < 512u) zrow[i] = 0u;                  // 2KB zero row (ND bf16)
}
__global__ void k_hist(const int* __restrict__ assign, unsigned* counts) {
    unsigned i = blockIdx.x * blockDim.x + threadIdx.x;
    if (i < NTOK) atomicAdd(&counts[assign[i] & (NE - 1)], 1u);
}
__global__ void k_scan(const unsigned* __restrict__ counts, unsigned* base9) {
    if (threadIdx.x == 0) {
        unsigned acc = 0;
        for (int e = 0; e < NE; ++e) {
            base9[e] = acc;
            acc += (counts[e] + 127u) & ~127u;   // pad segments to TILE_M
        }
        base9[NE] = acc;
    }
}
__global__ void k_scatter(const int* __restrict__ assign,
                          const unsigned* __restrict__ base9,
                          unsigned* cursors, unsigned* idx) {
    unsigned i = blockIdx.x * blockDim.x + threadIdx.x;
    if (i < NTOK) {
        unsigned e = assign[i] & (NE - 1);
        idx[base9[e] + atomicAdd(&cursors[e], 1u)] = i;
    }
}

// ------------------------------ prep ---------------------------------------
__global__ void k_cvt_x(const float* __restrict__ src, unsigned short* __restrict__ dst) {
    size_t i = ((size_t)blockIdx.x * blockDim.x + threadIdx.x) * 4u;
    float4 v = *(const float4*)(src + i);
    uint2 pk;
    pk.x = (unsigned)f2bf(v.x) | ((unsigned)f2bf(v.y) << 16);
    pk.y = (unsigned)f2bf(v.z) | ((unsigned)f2bf(v.w) << 16);
    *(uint2*)(dst + i) = pk;
}

// src f32 [R][C] (layer-strided) -> dst bf16 [C][R]
__global__ void k_cvt_T(const float* __restrict__ src, unsigned short* __restrict__ dst,
                        int R, int C) {
    __shared__ float tile[32][33];
    const size_t lay = (size_t)blockIdx.z * R * C;
    const float* s = src + lay;
    unsigned short* d = dst + lay;
    const int c0 = blockIdx.x * 32, r0 = blockIdx.y * 32;
    #pragma unroll
    for (int i = threadIdx.y; i < 32; i += 8)
        tile[i][threadIdx.x] = s[(size_t)(r0 + i) * C + c0 + threadIdx.x];
    __syncthreads();
    #pragma unroll
    for (int i = threadIdx.y; i < 32; i += 8)
        d[(size_t)(c0 + i) * R + r0 + threadIdx.x] = f2bf(tile[threadIdx.x][i]);
}

// --------------------- phase 1: x @ W_e[e] + b_e, relu ---------------------
// grid = (NH/TILE_N, MTILES), block = 256 (8 waves); wave tile 64x32
__global__ __launch_bounds__(256) void moe_ffn1(
    const unsigned short* __restrict__ xb, const unsigned short* __restrict__ WeT,
    const float* __restrict__ be, const unsigned* __restrict__ base9,
    const unsigned* __restrict__ idx, const unsigned short* __restrict__ zrow,
    unsigned short* __restrict__ hbuf)
{
    __shared__ __align__(16) unsigned char smA[2 * TILE_M * APITCH];
    __shared__ __align__(16) unsigned char smB[2 * TILE_N * BPITCH];

    const unsigned row0 = blockIdx.y * TILE_M;
    if (row0 >= base9[NE]) return;
    unsigned e = NE - 1;
    #pragma unroll
    for (int i = 0; i < NE; ++i) { if (row0 < base9[i + 1]) { e = (unsigned)i; break; } }

    const unsigned n0   = blockIdx.x * TILE_N;
    const unsigned tid  = threadIdx.x;
    const unsigned lane = tid & 31u, wave = tid >> 5;
    const unsigned m16  = lane & 15u, half = lane >> 4;
    const unsigned mrow = (wave & 1u) * 64u, ncol = (wave >> 1) * 32u;

    const unsigned sr = tid >> 1, sq = tid & 1u;           // staging row, 32B half
    const unsigned atok = idx[row0 + sr];
    const unsigned short* ga =
        ((atok != SENTINEL) ? (xb + (size_t)atok * ND) : zrow) + sq * 16u;
    const unsigned short* gb = WeT + ((size_t)e * NH + n0 + sr) * ND + sq * 16u;
    unsigned char* arow = smA + sr * APITCH + sq * 32u;
    unsigned char* brow = smB + sr * BPITCH + sq * 32u;

    v8f acc[4][2];
    #pragma unroll
    for (int mi = 0; mi < 4; ++mi)
        #pragma unroll
        for (int j = 0; j < 2; ++j)
            acc[mi][j] = (v8f){0.f,0.f,0.f,0.f,0.f,0.f,0.f,0.f};

    gemm_mainloop(ga, gb, ND, smA, smB, arow, brow, m16, half, mrow, ncol, acc);

    #pragma unroll
    for (int j = 0; j < 2; ++j) {
        const unsigned col = n0 + ncol + (unsigned)j * 16u + m16;
        const float bias = be[(size_t)e * NH + col];
        #pragma unroll
        for (int mi = 0; mi < 4; ++mi)
            #pragma unroll
            for (int i = 0; i < 8; ++i) {
                const unsigned rr = row0 + mrow + (unsigned)mi * 16u + (unsigned)i + half * 8u;
                hbuf[(size_t)rr * NH + col] = f2bf(fmaxf(acc[mi][j][i] + bias, 0.f));
            }
    }
}

// --------------------- phase 2: h @ W_out + b_out, scatter -----------------
// grid = (ND/TILE_N, MTILES), block = 256
__global__ __launch_bounds__(256) void moe_ffn2(
    const unsigned short* __restrict__ hbuf, const unsigned short* __restrict__ WoT,
    const float* __restrict__ bout, const unsigned* __restrict__ base9,
    const unsigned* __restrict__ idx, float* __restrict__ out)
{
    __shared__ __align__(16) unsigned char smA[2 * TILE_M * APITCH];
    __shared__ __align__(16) unsigned char smB[2 * TILE_N * BPITCH];

    const unsigned row0 = blockIdx.y * TILE_M;
    if (row0 >= base9[NE]) return;

    const unsigned n0   = blockIdx.x * TILE_N;
    const unsigned tid  = threadIdx.x;
    const unsigned lane = tid & 31u, wave = tid >> 5;
    const unsigned m16  = lane & 15u, half = lane >> 4;
    const unsigned mrow = (wave & 1u) * 64u, ncol = (wave >> 1) * 32u;

    const unsigned sr = tid >> 1, sq = tid & 1u;
    const unsigned short* ga = hbuf + (size_t)(row0 + sr) * NH + sq * 16u;
    const unsigned short* gb = WoT + (size_t)(n0 + sr) * NH + sq * 16u;
    unsigned char* arow = smA + sr * APITCH + sq * 32u;
    unsigned char* brow = smB + sr * BPITCH + sq * 32u;

    v8f acc[4][2];
    #pragma unroll
    for (int mi = 0; mi < 4; ++mi)
        #pragma unroll
        for (int j = 0; j < 2; ++j)
            acc[mi][j] = (v8f){0.f,0.f,0.f,0.f,0.f,0.f,0.f,0.f};

    gemm_mainloop(ga, gb, NH, smA, smB, arow, brow, m16, half, mrow, ncol, acc);

    #pragma unroll
    for (int j = 0; j < 2; ++j) {
        const unsigned col = n0 + ncol + (unsigned)j * 16u + m16;
        const float bias = bout[col];
        #pragma unroll
        for (int mi = 0; mi < 4; ++mi)
            #pragma unroll
            for (int i = 0; i < 8; ++i) {
                const unsigned rr = row0 + mrow + (unsigned)mi * 16u + (unsigned)i + half * 8u;
                const unsigned tok = idx[rr];
                if (tok != SENTINEL)
                    out[(size_t)tok * ND + col] = acc[mi][j][i] + bias;
            }
    }
}

// ------------------------------- launch ------------------------------------
extern "C" void kernel_launch(void* const* d_in, const int* in_sizes, int n_in,
                              void* d_out, int out_size, void* d_ws, size_t ws_size,
                              hipStream_t stream) {
    (void)in_sizes; (void)n_in; (void)out_size; (void)ws_size;
    const float* x    = (const float*)d_in[0];
    const float* We   = (const float*)d_in[1];
    const float* be   = (const float*)d_in[2];
    const float* Wout = (const float*)d_in[3];
    const float* bout = (const float*)d_in[4];
    const int* assign = (const int*)d_in[5];
    float* out = (float*)d_out;

    char* ws = (char*)d_ws;
    unsigned* counts  = (unsigned*)ws;
    unsigned* cursors = counts + 8;
    unsigned* base9   = counts + 16;
    unsigned* idx     = (unsigned*)(ws + IDX_OFF);
    unsigned* zrow32  = (unsigned*)(ws + ZROW_OFF);
    unsigned short* zrow = (unsigned short*)(ws + ZROW_OFF);
    unsigned short* xb   = (unsigned short*)(ws + XB_OFF);
    unsigned short* WeT  = (unsigned short*)(ws + WEB_OFF);
    unsigned short* WoT  = (unsigned short*)(ws + WOUTB_OFF);
    unsigned short* hbuf = (unsigned short*)(ws + H_OFF);

    k_init   <<<(MAXROWS + 255) / 256, 256, 0, stream>>>(counts, cursors, idx, zrow32);
    k_hist   <<<NTOK / 256, 256, 0, stream>>>(assign, counts);
    k_scan   <<<1, 32, 0, stream>>>(counts, base9);
    k_scatter<<<NTOK / 256, 256, 0, stream>>>(assign, base9, cursors, idx);

    k_cvt_x<<<(NTOK * (size_t)ND) / 4 / 256, 256, 0, stream>>>(x, xb);
    k_cvt_T<<<dim3(NH / 32, ND / 32, NE), dim3(32, 8), 0, stream>>>(We, WeT, ND, NH);
    k_cvt_T<<<dim3(ND / 32, NH / 32, 1),  dim3(32, 8), 0, stream>>>(Wout, WoT, NH, ND);

    moe_ffn1<<<dim3(NH / TILE_N, MTILES), 256, 0, stream>>>(xb, WeT, be, base9, idx, zrow, hbuf);
    moe_ffn2<<<dim3(ND / TILE_N, MTILES), 256, 0, stream>>>(hbuf, WoT, bout, base9, idx, out);
}
